// BiRNN_56736517980352
// MI455X (gfx1250) — matrix operand order repaired and checked by reading, not compiled, not run on hardware
//
#include <hip/hip_runtime.h>
#include <stdint.h>

// ---------------------------------------------------------------------------
// Bidirectional masked LSTM for MI455X (gfx1250), wave32 + WMMA bf16.
//   B=64, S=1024, D=512, U=512, G=4U=2048
// Phase 1: xz[dir] = bf16(x) @ bf16(W[dir]) + bias  (WMMA GEMM, 16x128/wave)
// Phase 2: sequential recurrence, one WG per (dir, 16-row batch tile),
//          h @ R via WMMA from LDS-resident bf16 h, gates fused in-register.
// WMMA kernels use waves_per_eu=1 so all 8 batched B-fragments stay live and
// the load clauses run ahead of the WMMA chain (one wait per K-step).
// ---------------------------------------------------------------------------

#define B_ 64
#define S_ 1024
#define D_ 512
#define U_ 512
#define G_ 2048

typedef unsigned short u16;
typedef __attribute__((ext_vector_type(16))) __bf16 bf16x16;
typedef __attribute__((ext_vector_type(8)))  float  v8f;

#define AS1 __attribute__((address_space(1)))
#define AS3 __attribute__((address_space(3)))

typedef int v4i __attribute__((vector_size(16)));   // matches builtin param type
typedef AS1 v4i* as1_v4i_p;
typedef AS3 v4i* as3_v4i_p;

#if __has_builtin(__builtin_amdgcn_global_load_async_to_lds_b128)
#define HAVE_ASYNC_LDS 1
#endif

struct Frag32 { uint4 lo; uint4 hi; };   // 32 bytes = 16 bf16 = 8 VGPRs

__device__ __forceinline__ bf16x16 make_frag(const void* p0, const void* p1) {
  Frag32 f;
  f.lo = *(const uint4*)p0;
  f.hi = *(const uint4*)p1;
  return __builtin_bit_cast(bf16x16, f);
}

__device__ __forceinline__ u16 f2bf(float x) {   // round-to-nearest-even
  unsigned int u = __builtin_bit_cast(unsigned int, x);
  u += 0x7FFFu + ((u >> 16) & 1u);
  return (u16)(u >> 16);
}

__device__ __forceinline__ float sigmoidf_(float x) { return 1.0f / (1.0f + __expf(-x)); }
__device__ __forceinline__ float tanhf_(float x)    { return 2.0f * sigmoidf_(2.0f * x) - 1.0f; }

__device__ __forceinline__ v8f wmma_bf16(bf16x16 a, bf16x16 b, v8f c) {
  return __builtin_amdgcn_wmma_f32_16x16x32_bf16(false, a, false, b, (short)0, c,
                                                 false, false);
}

// --------------------------- pack kernels ----------------------------------

__global__ __launch_bounds__(256) void pack_x_kernel(const float* __restrict__ x,
                                                     u16* __restrict__ xbf, int n4) {
  int i = blockIdx.x * 256 + threadIdx.x;
  if (i >= n4) return;
  float4 v = ((const float4*)x)[i];
  ushort4 o;
  o.x = f2bf(v.x); o.y = f2bf(v.y); o.z = f2bf(v.z); o.w = f2bf(v.w);
  ((ushort4*)xbf)[i] = o;
}

// Transpose [512 x 2048] f32 -> [2048 x 512] bf16 (coalesced writes).
__global__ __launch_bounds__(256) void pack_wt_kernel(
    const float* __restrict__ s0, const float* __restrict__ s1,
    const float* __restrict__ s2, const float* __restrict__ s3,
    u16* __restrict__ d0, u16* __restrict__ d1,
    u16* __restrict__ d2, u16* __restrict__ d3) {
  const float* src; u16* dst;
  switch (blockIdx.z) {
    case 0:  src = s0; dst = d0; break;
    case 1:  src = s1; dst = d1; break;
    case 2:  src = s2; dst = d2; break;
    default: src = s3; dst = d3; break;
  }
  int idx = blockIdx.x * 256 + threadIdx.x;      // 0 .. 2048*512-1
  int k = idx & (D_ - 1);
  int n = idx >> 9;
  dst[idx] = f2bf(src[(size_t)k * G_ + n]);      // dst[n][k] = src[k][n]
}

// --------------------------- phase 1: xz GEMM ------------------------------
// block = 256 threads (8 waves); wave w -> rows [bx*128 + 16w, +16),
// cols [by*128, +128). 8 accumulators/wave; B-frags batched per K-step.
// waves_per_eu=1: trade occupancy for live registers (deep load batching).
__global__ __launch_bounds__(256, 1) void gemm_xz_kernel(
    const u16* __restrict__ xbf,
    const u16* __restrict__ wtf, const u16* __restrict__ wtb,
    const float* __restrict__ biasf, const float* __restrict__ biasb,
    float* __restrict__ xzf, float* __restrict__ xzb) {
  const int dir = blockIdx.z;
  const u16*   wt   = dir ? wtb : wtf;
  const float* bias = dir ? biasb : biasf;
  float*       xz   = dir ? xzb : xzf;

  const int lane = threadIdx.x & 31;
  const int wave = threadIdx.x >> 5;
  const int ln = lane & 15, half = lane >> 4;
  const int mt = blockIdx.x * 128 + wave * 16;
  const int nb = blockIdx.y * 128;

  v8f acc[8] = {};
  const u16* aptr  = xbf + (size_t)(mt + ln) * D_ + half * 8;   // A-frag base
  const u16* bbase = wt + (size_t)(nb + ln) * D_ + half * 8;    // B-frag base

  bf16x16 a = make_frag(aptr, aptr + 16);
  for (int k0 = 0; k0 < D_; k0 += 32) {
    bf16x16 b[8];
#pragma unroll
    for (int j = 0; j < 8; ++j) {
      const u16* bp = bbase + (size_t)j * 16 * D_ + k0;
      b[j] = make_frag(bp, bp + 16);
    }
    __builtin_prefetch(aptr + k0 + 64, 0, 0);          // global_prefetch_b8
    bf16x16 an = a;
    if (k0 + 32 < D_) an = make_frag(aptr + k0 + 32, aptr + k0 + 48);
#pragma unroll
    for (int j = 0; j < 8; ++j) acc[j] = wmma_bf16(a, b[j], acc[j]);
    a = an;
  }
#pragma unroll
  for (int j = 0; j < 8; ++j) {
    const int col = nb + j * 16 + ln;
    const float bv = bias[col];
#pragma unroll
    for (int r = 0; r < 8; ++r) {
      const int row = mt + half * 8 + r;                // C/D layout: m = r + 8*half
      xz[(size_t)row * G_ + col] = acc[j][r] + bv;
    }
  }
}

// --------------------------- phase 2: recurrence ---------------------------
// grid = (B/16, 2); block = 512 threads = 16 waves; wave w owns u in [32w, 32w+32)
// and computes i/f/g/o gate columns for that slice (8 WMMA accumulators).
// Only 8 WGs exist -> occupancy is irrelevant; waves_per_eu=1 maximizes VGPRs.
__global__ __launch_bounds__(512, 1) void lstm_rec_kernel(
    const float* __restrict__ xzf, const float* __restrict__ xzb,
    const u16* __restrict__ rtf, const u16* __restrict__ rtb,
    const unsigned char* __restrict__ mask,
    float* __restrict__ out) {
  __shared__ u16 h_lds[16][U_ + 8];          // bf16 h tile, rows 16B-aligned (1040 B)
  __shared__ unsigned char mask_lds[16][S_]; // mask tile (16 KB)

  const int dir = blockIdx.y;
  const int bt  = blockIdx.x;
  const float* xz = dir ? xzb : xzf;
  const u16*   rt = dir ? rtb : rtf;

  const int tid  = threadIdx.x;
  const int lane = tid & 31;
  const int wave = tid >> 5;
  const int ln = lane & 15, half = lane >> 4;
  const int ub = wave * 32;

  for (int i = tid; i < 16 * (U_ + 8); i += 512) ((u16*)h_lds)[i] = 0;

  // stage mask tile into LDS (async DMA path when available)
#ifdef HAVE_ASYNC_LDS
  {
    const unsigned char* g = mask + (size_t)bt * 16 * S_;
#pragma unroll
    for (int it = 0; it < 2; ++it) {
      const int off = (tid + it * 512) * 16;           // 2 x 8KB = 16KB
      __builtin_amdgcn_global_load_async_to_lds_b128(
          (as1_v4i_p)(g + off),
          (as3_v4i_p)(((char*)mask_lds) + off), 0, 0);
    }
#if __has_builtin(__builtin_amdgcn_s_wait_asynccnt)
    __builtin_amdgcn_s_wait_asynccnt(0);
#else
    asm volatile("s_wait_asynccnt 0" ::: "memory");
#endif
  }
#else
  {
    const uint32_t* msrc = (const uint32_t*)(mask + (size_t)bt * 16 * S_);
    uint32_t* mdst = (uint32_t*)mask_lds;
    for (int i = tid; i < 16 * S_ / 4; i += 512) mdst[i] = msrc[i];
  }
#endif
  __syncthreads();

  v8f hreg[2] = {}, creg[2] = {};

  for (int s = 0; s < S_; ++s) {
    const int t = dir ? (S_ - 1 - s) : s;

    // ---- h @ R : 8 tiles (4 gates x 2 sub-tiles), A-frags from LDS ----
    v8f acc[8] = {};
    bf16x16 a = make_frag(&h_lds[ln][half * 8], &h_lds[ln][half * 8 + 16]);
    for (int k0 = 0; k0 < U_; k0 += 32) {
      bf16x16 b[8];
#pragma unroll
      for (int g = 0; g < 4; ++g) {
#pragma unroll
        for (int j = 0; j < 2; ++j) {
          const u16* bp =
              rt + (size_t)(g * U_ + ub + j * 16 + ln) * U_ + half * 8 + k0;
          b[g * 2 + j] = make_frag(bp, bp + 16);
        }
      }
      bf16x16 an = a;
      if (k0 + 32 < U_)
        an = make_frag(&h_lds[ln][half * 8 + k0 + 32],
                       &h_lds[ln][half * 8 + k0 + 48]);
#pragma unroll
      for (int i = 0; i < 8; ++i) acc[i] = wmma_bf16(a, b[i], acc[i]);
      a = an;
    }
    __syncthreads();  // all LDS h reads complete before overwriting h

    // ---- gates + state update + emit ----
#pragma unroll
    for (int j = 0; j < 2; ++j) {
#pragma unroll
      for (int r = 0; r < 8; ++r) {
        const int m = half * 8 + r;
        const int b = bt * 16 + m;
        const int u = ub + j * 16 + ln;
        const size_t rowoff = ((size_t)b * S_ + t) * G_;
        const float iz = acc[0 + j][r] + xz[rowoff + 0 * U_ + u];
        const float fz = acc[2 + j][r] + xz[rowoff + 1 * U_ + u];
        const float gz = acc[4 + j][r] + xz[rowoff + 2 * U_ + u];
        const float oz = acc[6 + j][r] + xz[rowoff + 3 * U_ + u];
        const float cn = sigmoidf_(fz) * creg[j][r] + sigmoidf_(iz) * tanhf_(gz);
        const float hn = sigmoidf_(oz) * tanhf_(cn);
        const bool mk = mask_lds[m][t] != 0;
        const float h2 = mk ? hn : hreg[j][r];
        const float c2 = mk ? cn : creg[j][r];
        hreg[j][r] = h2;
        creg[j][r] = c2;
        out[((size_t)b * S_ + t) * (2 * U_) + dir * U_ + u] = h2;
        h_lds[m][u] = f2bf(h2);
      }
    }
    __syncthreads();  // publish new h before next step's WMMA reads
  }

  // ---- final states: order = output, h_f, c_f, h_b, c_b ----
  const size_t outO = (size_t)B_ * S_ * 2 * U_;
  const size_t hO = outO + (size_t)dir * 2 * B_ * U_;
  const size_t cO = hO + (size_t)B_ * U_;
#pragma unroll
  for (int j = 0; j < 2; ++j) {
#pragma unroll
    for (int r = 0; r < 8; ++r) {
      const int m = half * 8 + r;
      const int b = bt * 16 + m;
      const int u = ub + j * 16 + ln;
      out[hO + (size_t)b * U_ + u] = hreg[j][r];
      out[cO + (size_t)b * U_ + u] = creg[j][r];
    }
  }
}

// --------------------------- launcher --------------------------------------
extern "C" void kernel_launch(void* const* d_in, const int* in_sizes, int n_in,
                              void* d_out, int out_size, void* d_ws, size_t ws_size,
                              hipStream_t stream) {
  const float* x  = (const float*)d_in[0];
  const unsigned char* mask = (const unsigned char*)d_in[1];  // jax bool_
  const float* kf = (const float*)d_in[2];
  const float* rf = (const float*)d_in[3];
  const float* bf = (const float*)d_in[4];
  const float* kb = (const float*)d_in[5];
  const float* rb = (const float*)d_in[6];
  const float* bb = (const float*)d_in[7];
  float* out = (float*)d_out;

  char* ws = (char*)d_ws;
  size_t o = 0;
  u16* xbf = (u16*)(ws + o); o += (size_t)B_ * S_ * D_ * 2;   // 64 MB
  u16* wtf = (u16*)(ws + o); o += (size_t)G_ * D_ * 2;        // 2 MB
  u16* rtf = (u16*)(ws + o); o += (size_t)G_ * U_ * 2;        // 2 MB
  u16* wtb = (u16*)(ws + o); o += (size_t)G_ * D_ * 2;        // 2 MB
  u16* rtb = (u16*)(ws + o); o += (size_t)G_ * U_ * 2;        // 2 MB
  float* xzf = (float*)(ws + o); o += (size_t)B_ * S_ * G_ * 4; // 512 MB
  float* xzb = (float*)(ws + o);                                // 512 MB

  pack_x_kernel<<<(B_ * S_ * D_ / 4 + 255) / 256, 256, 0, stream>>>(
      x, xbf, B_ * S_ * D_ / 4);

  dim3 gw((G_ * D_) / 256, 1, 4);
  pack_wt_kernel<<<gw, 256, 0, stream>>>(kf, rf, kb, rb, wtf, rtf, wtb, rtb);

  dim3 gg(B_ * S_ / 128, G_ / 128, 2);
  gemm_xz_kernel<<<gg, 256, 0, stream>>>(xbf, wtf, wtb, bf, bb, xzf, xzb);

  dim3 gr(B_ / 16, 2, 1);
  lstm_rec_kernel<<<gr, 512, 0, stream>>>(xzf, xzb, rtf, rtb, mask, out);
}
